// GraphAttentionEmbedding_3599182594724
// MI455X (gfx1250) — matrix operand order repaired and checked by reading, not compiled, not run on hardware
//
#include <hip/hip_runtime.h>
#include <hip/hip_bf16.h>
#include <stdint.h>

// ---------------- problem constants ----------------
#define NN      100000      // nodes
#define EE      800000      // edges
#define DIN     100         // in channels
#define HC      100         // H*C (heads * per-head dim)
#define CC      50          // per-head dim
#define EDGED   200         // edge_attr dim (time 100 | msg 100)
#define MT      (NN/16)     // 6250 node row tiles
#define KT_X    4           // K tiles for DIN=100 -> 128
#define NT_O    7           // N tiles for 100 -> 112
#define KT_E    7           // K tiles for EDGED=200 -> 224
#define EB      64          // edges per block in edge GEMM
#define INV_SQRT_C 0.14142135623730951f

typedef __attribute__((ext_vector_type(16))) __bf16 v16bf;
typedef __attribute__((ext_vector_type(8)))  float  v8f;
typedef __attribute__((ext_vector_type(4)))  int    v4i;

union ABf { uint32_t u[8]; uint4 q[2]; v16bf v; };

static __device__ __forceinline__ uint16_t f2bf(float f) {
    uint32_t u = __float_as_uint(f);
    uint32_t r = u + 0x7FFFu + ((u >> 16) & 1u);   // RNE
    return (uint16_t)(r >> 16);
}
static __device__ __forceinline__ uint32_t pack2(float lo, float hi) {
    return (uint32_t)f2bf(lo) | ((uint32_t)f2bf(hi) << 16);
}
static __device__ __forceinline__ v8f wmma_bf16(v16bf a, v16bf b, v8f c) {
    return __builtin_amdgcn_wmma_f32_16x16x32_bf16(false, a, false, b,
                                                   (short)0, c, false, false);
}
// monotonic float<->uint key for atomicMax-based segment max
static __device__ __forceinline__ uint32_t fkey(float f) {
    uint32_t b = __float_as_uint(f);
    return (b & 0x80000000u) ? ~b : (b | 0x80000000u);
}
static __device__ __forceinline__ float kinv(uint32_t k) {
    uint32_t b = (k & 0x80000000u) ? (k & 0x7FFFFFFFu) : ~k;
    return __uint_as_float(b);
}

// ---- CDNA5 async global->LDS copy (ASYNCcnt tracked) ----
// builtin param types (from hipcc diagnostic): (int4 AS1*, int4 AS3*, Imm, Imm)
static __device__ __forceinline__ void async_g2l_b128(const void* g, void* l) {
#if __has_builtin(__builtin_amdgcn_global_load_async_to_lds_b128)
    __builtin_amdgcn_global_load_async_to_lds_b128(
        (__attribute__((address_space(1))) v4i*)g,
        (__attribute__((address_space(3))) v4i*)l, 0, 0);
#else
    uint32_t loff = (uint32_t)(uintptr_t)(__attribute__((address_space(3))) char*)l;
    asm volatile("global_load_async_to_lds_b128 %0, %1, off"
                 :: "v"(loff), "v"((uint64_t)(uintptr_t)g) : "memory");
#endif
}
static __device__ __forceinline__ void wait_async0() {
#if __has_builtin(__builtin_amdgcn_s_wait_asynccnt)
    __builtin_amdgcn_s_wait_asynccnt(0);
#else
    asm volatile("s_wait_asynccnt 0x0" ::: "memory");
#endif
}

// ---------------- operand packing ----------------
// A-image: lane l holds row m=l&15; h=l>>4; VGPR v: K = v<4 ? h*8+2v : 8+h*8+2v
// B-image: lane l holds col n=l&15;          VGPR v: K = v<4 ? h*8+2v : 8+h*8+2v (same as A)
__global__ void pack_x_kernel(const float* __restrict__ x, uint32_t* __restrict__ xp) {
    int t = blockIdx.x * 256 + threadIdx.x;               // MT*KT_X*256 threads
    int tile = t >> 8, r = t & 255;
    int lane = r >> 3, v = r & 7;
    int mt = tile / KT_X, kt = tile % KT_X;
    int h = lane >> 4;
    int m = mt * 16 + (lane & 15);
    int K = kt * 32 + ((v < 4) ? (h * 8 + 2 * v) : (8 + h * 8 + 2 * v));
    float a = (K     < DIN) ? x[(size_t)m * DIN + K]     : 0.f;
    float b = (K + 1 < DIN) ? x[(size_t)m * DIN + K + 1] : 0.f;
    xp[t] = pack2(a, b);
}

__global__ void pack_b_kernel(const float* __restrict__ W, int K, int Nout,
                              int NT, int total, uint32_t* __restrict__ out) {
    int t = blockIdx.x * 256 + threadIdx.x;
    if (t >= total) return;
    int tile = t >> 8, r = t & 255;
    int lane = r >> 3, v = r & 7;
    int kt = tile / NT, nt = tile % NT;
    int n = nt * 16 + (lane & 15);
    int kb = kt * 32 + ((v < 4) ? ((lane >> 4) * 8 + 2 * v) : (8 + (lane >> 4) * 8 + 2 * v));
    float a = (kb     < K && n < Nout) ? W[(size_t)kb * Nout + n]       : 0.f;
    float b = (kb + 1 < K && n < Nout) ? W[(size_t)(kb + 1) * Nout + n] : 0.f;
    out[t] = pack2(a, b);
}

// ---------------- node-side GEMMs: q,k,v,skip = x@W + b ----------------
__global__ void node_gemm_kernel(const uint32_t* __restrict__ xp,
                                 const uint32_t* __restrict__ wq,
                                 const uint32_t* __restrict__ wk,
                                 const uint32_t* __restrict__ wv,
                                 const uint32_t* __restrict__ wsk,
                                 const float* __restrict__ bq,
                                 const float* __restrict__ bk,
                                 const float* __restrict__ bv,
                                 const float* __restrict__ bsk,
                                 float* __restrict__ qb, float* __restrict__ kb,
                                 float* __restrict__ vb, float* __restrict__ sb) {
    int wid  = blockIdx.x * 8 + (threadIdx.x >> 5);       // wave task id
    int lane = threadIdx.x & 31;
    int mt   = wid / 28;
    int task = wid % 28;
    int mat  = task / NT_O;
    int nt   = task % NT_O;

    const uint32_t* wp; const float* bias; float* ob;
    if (mat == 0)      { wp = wq;  bias = bq;  ob = qb; }
    else if (mat == 1) { wp = wk;  bias = bk;  ob = kb; }
    else if (mat == 2) { wp = wv;  bias = bv;  ob = vb; }
    else               { wp = wsk; bias = bsk; ob = sb; }

    v8f acc = {};
    #pragma unroll
    for (int kt = 0; kt < KT_X; ++kt) {
        ABf a, b;
        const uint4* ap = (const uint4*)(xp + (((size_t)(mt * KT_X + kt)) << 8) + lane * 8);
        a.q[0] = ap[0]; a.q[1] = ap[1];
        const uint4* bp = (const uint4*)(wp + (((size_t)(kt * NT_O + nt)) << 8) + lane * 8);
        b.q[0] = bp[0]; b.q[1] = bp[1];
        acc = wmma_bf16(a.v, b.v, acc);
    }
    int col = nt * 16 + (lane & 15);
    int h   = lane >> 4;
    if (col < HC) {
        float bval = bias[col];
        #pragma unroll
        for (int r = 0; r < 8; ++r) {
            int row = mt * 16 + r + 8 * h;                // C/D layout
            ob[(size_t)row * HC + col] = acc[r] + bval;
        }
    }
}

// ---------------- edge-side GEMM: e = [cos(rel*Wt+bt)|msg] @ We ----------------
// 128 threads (4 waves), 64 edges/block. msg is streamed global->LDS with
// CDNA5 async-to-LDS ops, overlapped with the cos() time-encoder packing.
__global__ void edge_gemm_kernel(const int* __restrict__ ei,       // [2,E]
                                 const float* __restrict__ last_update,
                                 const float* __restrict__ tvec,
                                 const float* __restrict__ msg,
                                 const float* __restrict__ Wt,
                                 const float* __restrict__ bt,
                                 const uint32_t* __restrict__ wep,
                                 float* __restrict__ ebuf) {
    __shared__ float    lds_msg[EB][DIN];                 // 25.6 KB raw f32 (async target)
    __shared__ uint32_t lds_cp[EB][CC];                   // 12.8 KB bf16 cos pairs
    __shared__ float    lds_rel[EB];
    __shared__ float    lds_wt[DIN];
    __shared__ float    lds_bt[DIN];

    int tid = threadIdx.x;
    int ebase = blockIdx.x * EB;

    if (tid == 0) __builtin_prefetch(wep, 0, 0);          // warm packed We

    // 1) launch async msg stream: EB rows * 400B = 25 x b128 per row
    for (int i = tid; i < EB * 25; i += 128) {
        int el = i / 25, c = i % 25;                      // c: 16B chunk index
        async_g2l_b128(msg + (size_t)(ebase + el) * DIN + c * 4,
                       &lds_msg[el][c * 4]);
    }

    // 2) stage rel_t / time-encoder params
    for (int j = tid; j < DIN; j += 128) { lds_wt[j] = Wt[j]; lds_bt[j] = bt[j]; }
    if (tid < EB) {
        int eg = ebase + tid;
        int s  = ei[eg];                                   // src
        lds_rel[tid] = last_update[s] - tvec[eg];
    }
    __syncthreads();

    // 3) cos() time-encoding -> bf16 pairs (overlaps the async msg stream)
    for (int i = tid; i < EB * CC; i += 128) {
        int el = i / CC, p = i - el * CC;
        int K  = 2 * p;
        float rel = lds_rel[el];
        float v0 = __cosf(fmaf(rel, lds_wt[K],     lds_bt[K]));
        float v1 = __cosf(fmaf(rel, lds_wt[K + 1], lds_bt[K + 1]));
        lds_cp[el][p] = pack2(v0, v1);
    }

    wait_async0();                                         // own wave's ASYNCcnt -> 0
    __syncthreads();                                       // all waves' data visible

    // 4) per-wave 16x(224)x112 WMMA sweep
    int w    = tid >> 5;                                   // wave -> 16 edges
    int lane = tid & 31;
    int h    = lane >> 4;
    int m    = w * 16 + (lane & 15);

    ABf a[KT_E];
    #pragma unroll
    for (int kt = 0; kt < KT_E; ++kt)
        #pragma unroll
        for (int v = 0; v < 8; ++v) {
            int K = kt * 32 + ((v < 4) ? (h * 8 + 2 * v) : (8 + h * 8 + 2 * v));
            uint32_t pw;
            if (K < DIN)         pw = lds_cp[m][K >> 1];
            else if (K < EDGED)  pw = pack2(lds_msg[m][K - DIN], lds_msg[m][K - DIN + 1]);
            else                 pw = 0u;
            a[kt].u[v] = pw;
        }

    #pragma unroll
    for (int nt = 0; nt < NT_O; ++nt) {
        v8f acc = {};
        #pragma unroll
        for (int kt = 0; kt < KT_E; ++kt) {
            ABf b;
            const uint4* bp = (const uint4*)(wep + (((size_t)(kt * NT_O + nt)) << 8) + lane * 8);
            b.q[0] = bp[0]; b.q[1] = bp[1];
            acc = wmma_bf16(a[kt].v, b.v, acc);
        }
        int col = nt * 16 + (lane & 15);
        if (col < HC) {
            #pragma unroll
            for (int r = 0; r < 8; ++r) {
                int row = ebase + w * 16 + r + 8 * h;
                ebuf[(size_t)row * HC + col] = acc[r];
            }
        }
    }
}

// ---------------- attention: logits + segment max ----------------
__global__ void att_logits_kernel(const int* __restrict__ ei,
                                  const float* __restrict__ qb,
                                  const float* __restrict__ kb,
                                  const float* __restrict__ ebuf,
                                  float* __restrict__ logit,
                                  uint32_t* __restrict__ amaxk) {
    int e = blockIdx.x * 256 + threadIdx.x;
    if (e >= EE) return;
    int s = ei[e], d = ei[EE + e];
    const float* qr = qb   + (size_t)d * HC;
    const float* kr = kb   + (size_t)s * HC;
    const float* er = ebuf + (size_t)e * HC;
    float d0 = 0.f, d1 = 0.f;
    #pragma unroll 10
    for (int c = 0; c < CC; ++c)       d0 = fmaf(qr[c], kr[c] + er[c], d0);
    #pragma unroll 10
    for (int c = CC; c < HC; ++c)      d1 = fmaf(qr[c], kr[c] + er[c], d1);
    d0 *= INV_SQRT_C; d1 *= INV_SQRT_C;
    logit[(size_t)e * 2 + 0] = d0;
    logit[(size_t)e * 2 + 1] = d1;
    atomicMax(&amaxk[d * 2 + 0], fkey(d0));
    atomicMax(&amaxk[d * 2 + 1], fkey(d1));
}

// ---------------- attention: exp weights + segment sum ----------------
__global__ void att_weights_kernel(const int* __restrict__ ei,
                                   const float* __restrict__ logit,
                                   const uint32_t* __restrict__ amaxk,
                                   float* __restrict__ wbuf,
                                   float* __restrict__ asum) {
    int e = blockIdx.x * 256 + threadIdx.x;
    if (e >= EE) return;
    int d = ei[EE + e];
    #pragma unroll
    for (int h = 0; h < 2; ++h) {
        float mx = kinv(amaxk[d * 2 + h]);
        float w  = __expf(logit[(size_t)e * 2 + h] - mx);
        wbuf[(size_t)e * 2 + h] = w;
        atomicAdd(&asum[d * 2 + h], w);
    }
}

// ---------------- attention: weighted scatter-accumulate ----------------
__global__ void att_accum_kernel(const int* __restrict__ ei,
                                 const float* __restrict__ wbuf,
                                 const float* __restrict__ vb,
                                 const float* __restrict__ ebuf,
                                 float* __restrict__ accum) {
    long long i = (long long)blockIdx.x * 256 + threadIdx.x;   // E*HC threads
    int e  = (int)(i / HC);
    int hc = (int)(i - (long long)e * HC);
    int s = ei[e], d = ei[EE + e];
    float w = wbuf[(size_t)e * 2 + (hc >= CC ? 1 : 0)];
    float val = w * (vb[(size_t)s * HC + hc] + ebuf[i]);
    atomicAdd(&accum[(size_t)d * HC + hc], val);
}

// ---------------- finalize: normalize + skip ----------------
__global__ void finalize_kernel(const float* __restrict__ accum,
                                const float* __restrict__ asum,
                                const float* __restrict__ sb,
                                float* __restrict__ out) {
    long long i = (long long)blockIdx.x * 256 + threadIdx.x;
    if (i >= (long long)NN * HC) return;
    int n  = (int)(i / HC);
    int hc = (int)(i - (long long)n * HC);
    float denom = asum[n * 2 + (hc >= CC ? 1 : 0)] + 1e-16f;
    out[i] = accum[i] / denom + sb[i];
}

// ---------------- host launcher ----------------
extern "C" void kernel_launch(void* const* d_in, const int* in_sizes, int n_in,
                              void* d_out, int out_size, void* d_ws, size_t ws_size,
                              hipStream_t stream) {
    const float* x     = (const float*)d_in[0];
    const float* lu    = (const float*)d_in[1];
    const int*   ei    = (const int*)  d_in[2];
    const float* tv    = (const float*)d_in[3];
    const float* msg   = (const float*)d_in[4];
    const float* Wt    = (const float*)d_in[5];
    const float* bt    = (const float*)d_in[6];
    const float* Wq    = (const float*)d_in[7];
    const float* bq    = (const float*)d_in[8];
    const float* Wk    = (const float*)d_in[9];
    const float* bk    = (const float*)d_in[10];
    const float* Wv    = (const float*)d_in[11];
    const float* bv    = (const float*)d_in[12];
    const float* We    = (const float*)d_in[13];
    const float* Wsk   = (const float*)d_in[14];
    const float* bsk   = (const float*)d_in[15];
    float* out = (float*)d_out;

    uint8_t* p = (uint8_t*)d_ws;
    auto take = [&](size_t bytes) -> void* {
        void* r = (void*)p; p += (bytes + 255) & ~(size_t)255; return r;
    };
    uint32_t* xp    = (uint32_t*)take((size_t)MT * KT_X * 256 * 4);     // 25.6 MB
    uint32_t* wqp   = (uint32_t*)take((size_t)KT_X * NT_O * 256 * 4);
    uint32_t* wkp   = (uint32_t*)take((size_t)KT_X * NT_O * 256 * 4);
    uint32_t* wvp   = (uint32_t*)take((size_t)KT_X * NT_O * 256 * 4);
    uint32_t* wsp   = (uint32_t*)take((size_t)KT_X * NT_O * 256 * 4);
    uint32_t* wep   = (uint32_t*)take((size_t)KT_E * NT_O * 256 * 4);
    float* qb    = (float*)take((size_t)NN * HC * 4);                   // 40 MB each
    float* kb    = (float*)take((size_t)NN * HC * 4);
    float* vb    = (float*)take((size_t)NN * HC * 4);
    float* sb    = (float*)take((size_t)NN * HC * 4);
    float* ebuf  = (float*)take((size_t)EE * HC * 4);                   // 320 MB
    float* logit = (float*)take((size_t)EE * 2 * 4);
    float* wbuf  = (float*)take((size_t)EE * 2 * 4);
    uint32_t* amaxk = (uint32_t*)take((size_t)NN * 2 * 4);
    float* asum  = (float*)take((size_t)NN * 2 * 4);
    float* accum = (float*)take((size_t)NN * HC * 4);
    (void)ws_size; (void)n_in; (void)in_sizes; (void)out_size;

    (void)hipMemsetAsync(amaxk, 0, (size_t)NN * 2 * 4, stream);  // key 0 < key(-inf)
    (void)hipMemsetAsync(asum,  0, (size_t)NN * 2 * 4, stream);
    (void)hipMemsetAsync(accum, 0, (size_t)NN * HC * 4, stream);

    // operand packing
    pack_x_kernel<<<MT * KT_X, 256, 0, stream>>>(x, xp);
    pack_b_kernel<<<KT_X * NT_O, 256, 0, stream>>>(Wq,  DIN,   HC, NT_O, KT_X * NT_O * 256, wqp);
    pack_b_kernel<<<KT_X * NT_O, 256, 0, stream>>>(Wk,  DIN,   HC, NT_O, KT_X * NT_O * 256, wkp);
    pack_b_kernel<<<KT_X * NT_O, 256, 0, stream>>>(Wv,  DIN,   HC, NT_O, KT_X * NT_O * 256, wvp);
    pack_b_kernel<<<KT_X * NT_O, 256, 0, stream>>>(Wsk, DIN,   HC, NT_O, KT_X * NT_O * 256, wsp);
    pack_b_kernel<<<KT_E * NT_O, 256, 0, stream>>>(We,  EDGED, HC, NT_O, KT_E * NT_O * 256, wep);

    // WMMA GEMMs
    node_gemm_kernel<<<(MT * 28) / 8, 256, 0, stream>>>(xp, wqp, wkp, wvp, wsp,
                                                        bq, bk, bv, bsk, qb, kb, vb, sb);
    edge_gemm_kernel<<<EE / EB, 128, 0, stream>>>(ei, lu, tv, msg, Wt, bt, wep, ebuf);

    // attention / segment softmax / aggregation
    att_logits_kernel <<<EE / 256, 256, 0, stream>>>(ei, qb, kb, ebuf, logit, amaxk);
    att_weights_kernel<<<EE / 256, 256, 0, stream>>>(ei, logit, amaxk, wbuf, asum);
    att_accum_kernel  <<<(EE * HC) / 256, 256, 0, stream>>>(ei, wbuf, vb, ebuf, accum);
    finalize_kernel   <<<((size_t)NN * HC + 255) / 256, 256, 0, stream>>>(accum, asum, sb, out);
}
